// DiagonalBiLSTM_35751307772247
// MI455X (gfx1250) — matrix operand order repaired and checked by reading, not compiled, not run on hardware
//
#include <hip/hip_runtime.h>
#include <hip/hip_bf16.h>
#include <math.h>

typedef __bf16 bf16_t;
typedef __bf16 v16bf __attribute__((ext_vector_type(16)));
typedef __bf16 v8bf  __attribute__((ext_vector_type(8)));
typedef float  v8f   __attribute__((ext_vector_type(8)));

namespace {

constexpr int kB = 16, kC = 256, kG = 1024, kH = 64, kW = 64;
constexpr int kWD = 2 * kW - 1;   // 127 diagonal steps
constexpr int kR  = kB * kH;      // 1024 rows (b*h flattened)
constexpr int kRT = 16;           // rows per tile (one WMMA M)
constexpr int kNWG = kR / kRT;    // 64 persistent workgroups
constexpr int kHS = kC + 8;       // LDS bf16 row stride (16B aligned, bank spread)
constexpr int kGS = kG + 8;       // LDS bf16 row stride for gates
constexpr int kTS = kW + 1;       // xpose LDS stride (odd -> conflict-free)

__device__ __forceinline__ v16bf cat8(v8bf lo, v8bf hi) {
  return __builtin_shufflevector(lo, hi, 0, 1, 2, 3, 4, 5, 6, 7,
                                 8, 9, 10, 11, 12, 13, 14, 15);
}

__device__ __forceinline__ v8f wmma_bf16(v16bf a, v16bf b, v8f c) {
  // D = A(16x32) * B(32x16) + C, f32 accumulate
  return __builtin_amdgcn_wmma_f32_16x16x32_bf16(
      false, a, false, b, (short)0, c, false, false);
}

// B-fragment: weight stored [N][256] row-major (K contiguous), B[k][n] = W[n][k].
// Lane: n = lane&15; K = (lane<16 ? 0..7,16..23 : 8..15,24..31).
__device__ __forceinline__ v16bf ldB(const bf16_t* __restrict__ W, int n0,
                                     int kc, int lane) {
  const bf16_t* p =
      W + (size_t)(n0 + (lane & 15)) * kC + kc + ((lane & 16) ? 8 : 0);
  return cat8(*(const v8bf*)p, *(const v8bf*)(p + 16));
}

__device__ __forceinline__ float sigm(float x) {
  return 1.0f / (1.0f + __expf(-x));
}

// ---------------- prep: bf16 weights, fused bias, zero state ----------------
__global__ __launch_bounds__(256) void prep_kernel(
    const float* __restrict__ w_i2s, const float* __restrict__ w_ih,
    const float* __restrict__ b_ih, const float* __restrict__ b_hh,
    const float* __restrict__ k0, const float* __restrict__ k1,
    bf16_t* __restrict__ wm, bf16_t* __restrict__ wihb,
    bf16_t* __restrict__ k0b, bf16_t* __restrict__ k1b,
    float* __restrict__ bias_g, bf16_t* __restrict__ h0,
    bf16_t* __restrict__ h1, float* __restrict__ cbuf,
    bf16_t* __restrict__ zpix, unsigned* __restrict__ counters) {
  unsigned gid = blockIdx.x * 256u + threadIdx.x;
  if (gid < 65536u) {  // 256x256 weights; fold MaskedConv 'B' mask into w_i2s
    unsigned o = gid >> 8, c = gid & 255u;
    float msk = ((c % 3u) <= (o % 3u)) ? 1.0f : 0.0f;
    wm[gid]  = (bf16_t)(w_i2s[gid] * msk);
    k0b[gid] = (bf16_t)k0[gid];
    k1b[gid] = (bf16_t)k1[gid];
  }
  if (gid < 262144u) {  // 1024x256
    wihb[gid] = (bf16_t)w_ih[gid];
    cbuf[gid] = 0.0f;
  }
  if (gid < 1024u) bias_g[gid] = b_ih[gid] + b_hh[gid];
  if (gid < 266240u) {  // (1024+16) x 256, zero pad rows for h_next shift
    h0[gid] = (bf16_t)0.0f;
    h1[gid] = (bf16_t)0.0f;
  }
  if (gid < 256u) zpix[gid] = (bf16_t)0.0f;  // zero A-row for skew borders
  if (gid < 16u) counters[gid] = 0u;
}

// reset recurrent state between the two directional scans
__global__ __launch_bounds__(256) void reset_kernel(bf16_t* __restrict__ h0,
                                                    bf16_t* __restrict__ h1,
                                                    float* __restrict__ cbuf) {
  unsigned gid = blockIdx.x * 256u + threadIdx.x;
  if (gid < 262144u) cbuf[gid] = 0.0f;
  if (gid < 266240u) {
    h0[gid] = (bf16_t)0.0f;
    h1[gid] = (bf16_t)0.0f;
  }
}

// ------- x [B][C][H][W] f32 -> xt [B][H][W][C] bf16, LDS-tiled -------
// One block per (b,h): coalesced f32 reads along w, coalesced bf16 writes
// along c, odd LDS stride for conflict-free transpose.
__global__ __launch_bounds__(256) void xpose_kernel(const float* __restrict__ x,
                                                    bf16_t* __restrict__ xt) {
  __shared__ bf16_t tile[kC * kTS];
  const int bh = blockIdx.x;        // b*64 + h
  const int b  = bh >> 6;
  const int h  = bh & 63;
  const int tid = threadIdx.x;
  const float* src = x + (((size_t)b * kC) * kH + h) * kW;  // + c*H*W + w
  for (int i = 0; i < kC * kW / 256; ++i) {
    int idx = tid + i * 256;
    int c = idx >> 6, w = idx & 63;
    tile[c * kTS + w] = (bf16_t)src[(size_t)c * kH * kW + w];
  }
  __syncthreads();
  bf16_t* dst = xt + ((size_t)bh * kW << 8);  // [w][c]
  for (int i = 0; i < kC * kW / 256; ++i) {
    int idx = tid + i * 256;
    int w = idx >> 8, c = idx & 255;
    dst[((size_t)w << 8) + c] = tile[c * kTS + w];
  }
}

// -------- skewed masked 1x1 conv: xs[dir][t][r][c], both directions --------
__global__ __launch_bounds__(256) void i2s_kernel(
    const bf16_t* __restrict__ xt, const bf16_t* __restrict__ wm,
    const float* __restrict__ b_i2s, const bf16_t* __restrict__ zpix,
    bf16_t* __restrict__ xs_l, bf16_t* __restrict__ xs_r) {
  const int t   = blockIdx.y;        // diagonal step
  const int r0  = blockIdx.x * kRT;  // row tile
  const int dir = blockIdx.z;        // 0 = left->right, 1 = flipped
  bf16_t* __restrict__ xs = dir ? xs_r : xs_l;
  const int lane = threadIdx.x & 31;
  const int wave = threadIdx.x >> 5;

  // A-fragment row for this lane: one skewed pixel (channel-contiguous).
  // Out-of-band rows read from the zero pixel -> unconditional loads.
  const int m    = lane & 15;
  const int r    = r0 + m;
  const int bq   = r >> 6;
  const int hrow = r & 63;
  const int wcol = t - hrow;  // skew: row h shifted right by h
  const bool valid = (wcol >= 0) && (wcol < kW);
  const int wcell  = dir ? (kW - 1 - wcol) : wcol;
  const bf16_t* apix =
      valid ? xt + ((((size_t)bq * kH + hrow) * kW + wcell) << 8) : zpix;
  const int kbA = (lane & 16) ? 8 : 0;
  const int nlo = lane & 15;
  const int mh  = (lane & 16) ? 8 : 0;

  for (int nt = 0; nt < 2; ++nt) {  // 8 waves x 2 = 16 n-tiles over C=256
    const int n0 = (wave * 2 + nt) * 16;
    v8f acc = {};
    for (int kc = 0; kc < kC; kc += 32) {
      v16bf a = cat8(*(const v8bf*)(apix + kc + kbA),
                     *(const v8bf*)(apix + kc + kbA + 16));
      acc = wmma_bf16(a, ldB(wm, n0, kc, lane), acc);
    }
    const float bi = b_i2s[n0 + nlo];
    bf16_t* orow = xs + ((size_t)t * kR + r0) * kC + n0 + nlo;
    for (int v = 0; v < 8; ++v)
      orow[(size_t)(v + mh) * kC] = (bf16_t)(acc[v] + bi);
  }
}

// ---------------- persistent diagonal scan (one direction) ----------------
__global__ __launch_bounds__(256) void scan_kernel(
    const bf16_t* __restrict__ xs, const bf16_t* __restrict__ k0w,
    const bf16_t* __restrict__ k1w, const bf16_t* __restrict__ wih,
    const float* __restrict__ b_s2s, const float* __restrict__ bias_g,
    bf16_t* __restrict__ hb0, bf16_t* __restrict__ hb1,
    float* __restrict__ cbuf, float* __restrict__ out,
    unsigned* __restrict__ counter, int dir) {
  __shared__ bf16_t sh_h[(kRT + 1) * kHS];  // h rows r0..r0+16
  __shared__ bf16_t sh_u[kRT * kHS];        // x_t + s2s, bf16
  __shared__ bf16_t sh_g[kRT * kGS];        // gate pre-activations

  const int tid  = threadIdx.x;
  const int lane = tid & 31;
  const int wave = tid >> 5;
  const int r0   = blockIdx.x * kRT;
  const int arow = lane & 15;
  const int kbA  = (lane & 16) ? 8 : 0;
  const int nlo  = lane & 15;
  const int mh   = (lane & 16) ? 8 : 0;

  for (int t = 0; t < kWD; ++t) {
    const bf16_t* hcur = (t & 1) ? hb1 : hb0;
    bf16_t*       hnxt = (t & 1) ? hb0 : hb1;
    const bf16_t* xsp  = xs + ((size_t)t * kR + r0) * kC;

    if (t + 1 < kWD)  // pull next step's skewed inputs toward L2/WGP$
      __builtin_prefetch(xs + ((size_t)(t + 1) * kR + r0) * kC, 0, 1);

    // stage h_prev rows r0..r0+16 in 16B chunks (row r0+16 from neighbor tile)
    for (int idx = tid; idx < (kRT + 1) * (kC / 8); idx += 256) {
      int rr = idx >> 5, cc = (idx & 31) * 8;
      *(v8bf*)&sh_h[rr * kHS + cc] =
          *(const v8bf*)&hcur[(size_t)(r0 + rr) * kC + cc];
    }
    __syncthreads();

    // ---- s2s = h@k0^T + h_shift@k1^T  (+ b_s2s + x_t) -> u ----
    for (int nt = 0; nt < 2; ++nt) {
      const int n0 = (wave * 2 + nt) * 16;
      v8f acc = {};
      for (int kc = 0; kc < kC; kc += 32) {
        const bf16_t* p0 = &sh_h[arow * kHS + kc + kbA];
        const bf16_t* p1 = &sh_h[(arow + 1) * kHS + kc + kbA];
        v16bf a0 = cat8(*(const v8bf*)p0, *(const v8bf*)(p0 + 16));
        v16bf a1 = cat8(*(const v8bf*)p1, *(const v8bf*)(p1 + 16));
        acc = wmma_bf16(a0, ldB(k0w, n0, kc, lane), acc);
        acc = wmma_bf16(a1, ldB(k1w, n0, kc, lane), acc);
      }
      const float bs = b_s2s[n0 + nlo];
      for (int v = 0; v < 8; ++v) {
        const int mm = v + mh;
        float u = acc[v] + bs + (float)xsp[(size_t)mm * kC + n0 + nlo];
        sh_u[mm * kHS + n0 + nlo] = (bf16_t)u;
      }
    }
    __syncthreads();

    // ---- gates = u @ w_ih^T + bias  ([16,256] x [256,1024]) ----
    for (int nt = 0; nt < 8; ++nt) {
      const int n0 = (wave * 8 + nt) * 16;
      v8f acc = {};
      for (int kc = 0; kc < kC; kc += 32) {
        const bf16_t* ap = &sh_u[arow * kHS + kc + kbA];
        v16bf a = cat8(*(const v8bf*)ap, *(const v8bf*)(ap + 16));
        acc = wmma_bf16(a, ldB(wih, n0, kc, lane), acc);
      }
      const float bg = bias_g[n0 + nlo];
      for (int v = 0; v < 8; ++v)
        sh_g[(v + mh) * kGS + n0 + nlo] = (bf16_t)(acc[v] + bg);
    }
    __syncthreads();

    // ---- LSTM cell + direct scatter into final output ----
    for (int e = tid; e < kRT * kC; e += 256) {
      const int mm = e >> 8;
      const int co = e & 255;
      const float gi = (float)sh_g[mm * kGS + co];
      const float gf = (float)sh_g[mm * kGS + 256 + co];
      const float gg = (float)sh_g[mm * kGS + 512 + co];
      const float go = (float)sh_g[mm * kGS + 768 + co];
      const size_t ci = (size_t)(r0 + mm) * kC + co;
      float cn = sigm(gf) * cbuf[ci] + sigm(gi) * tanhf(gg);
      float hn = sigm(go) * tanhf(cn);
      cbuf[ci] = cn;
      hnxt[ci] = (bf16_t)hn;

      // faithful reshape (wd, b*h*C) -> (wd, b, C, h), then unshift
      const unsigned f = (unsigned)(r0 + mm) * 256u + (unsigned)co;
      const int bb = f >> 14;
      const int cO = (f >> 6) & 255;
      const int ii = f & 63;
      const int jj = t - ii;
      if (jj >= 0 && jj < kW) {
        if (dir == 0) {
          out[((((size_t)bb * kC + cO) * kH) + ii) * kW + jj] = hn;
        } else if (ii + 1 < kH) {  // flipped + shifted down one row
          out[((((size_t)bb * kC + cO) * kH) + ii + 1) * kW + (kW - 1 - jj)] +=
              hn;
        }
      }
    }

    // ---- device-wide step barrier (release/acquire around atomics) ----
    __threadfence();
    __syncthreads();
    if (tid == 0) {
      atomicAdd(counter, 1u);
      const unsigned target = (unsigned)(t + 1) * (unsigned)kNWG;
      while (atomicAdd(counter, 0u) < target) {
        __builtin_amdgcn_s_sleep(1);
      }
    }
    __syncthreads();
    __threadfence();
  }
}

}  // namespace

extern "C" void kernel_launch(void* const* d_in, const int* in_sizes, int n_in,
                              void* d_out, int out_size, void* d_ws,
                              size_t ws_size, hipStream_t stream) {
  const float* x     = (const float*)d_in[0];
  const float* w_i2s = (const float*)d_in[1];
  const float* b_i2s = (const float*)d_in[2];
  const float* w_ih  = (const float*)d_in[3];
  const float* b_ih  = (const float*)d_in[4];
  const float* b_hh  = (const float*)d_in[5];
  const float* k0    = (const float*)d_in[6];
  const float* k1    = (const float*)d_in[7];
  const float* b_s2s = (const float*)d_in[8];
  float* out = (float*)d_out;
  (void)in_sizes; (void)n_in; (void)out_size; (void)ws_size;

  char* ws = (char*)d_ws;
  size_t off = 0;
  auto take = [&](size_t bytes) -> char* {
    char* p = ws + off;
    off += (bytes + 255) & ~(size_t)255;
    return p;
  };

  bf16_t* xt   = (bf16_t*)take((size_t)kB * kH * kW * kC * 2);   // 32 MB
  bf16_t* xs_l = (bf16_t*)take((size_t)kWD * kR * kC * 2);       // 64 MB
  bf16_t* xs_r = (bf16_t*)take((size_t)kWD * kR * kC * 2);       // 64 MB
  bf16_t* wm   = (bf16_t*)take((size_t)kC * kC * 2);
  bf16_t* k0b  = (bf16_t*)take((size_t)kC * kC * 2);
  bf16_t* k1b  = (bf16_t*)take((size_t)kC * kC * 2);
  bf16_t* wihb = (bf16_t*)take((size_t)kG * kC * 2);
  float*  biag = (float*)take((size_t)kG * 4);
  bf16_t* h0   = (bf16_t*)take((size_t)(kR + kRT) * kC * 2);
  bf16_t* h1   = (bf16_t*)take((size_t)(kR + kRT) * kC * 2);
  float*  cbuf = (float*)take((size_t)kR * kC * 4);
  bf16_t* zpix = (bf16_t*)take((size_t)kC * 2);
  unsigned* ctr = (unsigned*)take(256);

  prep_kernel<<<1040, 256, 0, stream>>>(w_i2s, w_ih, b_ih, b_hh, k0, k1, wm,
                                        wihb, k0b, k1b, biag, h0, h1, cbuf,
                                        zpix, ctr);
  xpose_kernel<<<kB * kH, 256, 0, stream>>>(x, xt);
  i2s_kernel<<<dim3(kNWG, kWD, 2), 256, 0, stream>>>(xt, wm, b_i2s, zpix,
                                                     xs_l, xs_r);
  scan_kernel<<<kNWG, 256, 0, stream>>>(xs_l, k0b, k1b, wihb, b_s2s, biag, h0,
                                        h1, cbuf, out, &ctr[0], 0);
  reset_kernel<<<1040, 256, 0, stream>>>(h0, h1, cbuf);
  scan_kernel<<<kNWG, 256, 0, stream>>>(xs_r, k0b, k1b, wihb, b_s2s, biag, h0,
                                        h1, cbuf, out, &ctr[4], 1);
}
